// xLSTMRegressorUnified_77506979824206
// MI455X (gfx1250) — compile-verified, matrix operand order checked by reading, not run on hardware
//
#include <hip/hip_runtime.h>
#include <hip/hip_bf16.h>
#include <cstdint>
#include <cstddef>

// ---------------- model constants ----------------
constexpr int kB   = 512;
constexpr int kT   = 32;
constexpr int kFPS = 64;
constexpr int kD   = 512;
constexpr int kI   = 1024;
constexpr int kNH  = 8;
constexpr int kNB  = 256;
constexpr int kKc  = 4;     // conv taps
constexpr int kL   = 4;
constexpr int kDH  = kI / kNH;           // 128
constexpr long long kNT = (long long)kB * kT;  // 16384 tokens

// ---------------- WMMA types ----------------
typedef __attribute__((ext_vector_type(16))) __bf16 v16bf;
typedef __attribute__((ext_vector_type(8)))  float  v8f;

struct Frag32B { int4 lo, hi; };   // 32 bytes == v16bf

__device__ __forceinline__ uint16_t f2bf(float f) {
    uint32_t u = __builtin_bit_cast(uint32_t, f);
    uint32_t r = u + 0x7FFFu + ((u >> 16) & 1u);   // RNE
    return (uint16_t)(r >> 16);
}

__device__ __forceinline__ float swishf(float x) {
    return x / (1.f + expf(-x));
}

__device__ __forceinline__ v16bf make_frag(const uint32_t* lds, int row, int dwBase) {
    Frag32B f;
    f.lo = *reinterpret_cast<const int4*>(lds + row * 20 + dwBase);
    f.hi = *reinterpret_cast<const int4*>(lds + row * 20 + dwBase + 8);
    return __builtin_bit_cast(v16bf, f);
}

// ---- async global->LDS staging (ASYNCcnt path), device-pass only, builtin-gated ----
#if defined(__HIP_DEVICE_COMPILE__) && \
    __has_builtin(__builtin_amdgcn_global_load_async_to_lds_b128) && \
    __has_builtin(__builtin_amdgcn_s_wait_asynccnt)
#define USE_ASYNC_LDS 1
#else
#define USE_ASYNC_LDS 0
#endif

#if USE_ASYNC_LDS
typedef int v4i_raw __attribute__((vector_size(16)));   // builtin wants raw vector pointee
__device__ __forceinline__ void async_copy16(const void* g, void* l) {
    // global addresses are value-preserving in AS1; LDS offset = low 32 bits of the
    // generic shared pointer (ISA: flat->LDS truncates to addr[31:0]).
    __attribute__((address_space(1))) v4i_raw* gp =
        (__attribute__((address_space(1))) v4i_raw*)(unsigned long long)(uintptr_t)g;
    __attribute__((address_space(3))) v4i_raw* lp =
        (__attribute__((address_space(3))) v4i_raw*)(unsigned)(uintptr_t)l;
    __builtin_amdgcn_global_load_async_to_lds_b128(gp, lp, 0, 0);
}
#endif

// ---------------- generic bf16 WMMA GEMM: C[M,N] = A[M,K] * Bw[N,K]^T (+bias)(+=C) ----------------
// 256 threads = 8 waves. Wave grid WGM x WGN, each wave computes WTM x WTN tiles of 16x16.
// Double-buffered LDS; async DMA staging when the toolchain exposes the gfx1250 builtins.
template<int BM, int BN, int WGM, int WGN, int WTM, int WTN>
__global__ __launch_bounds__(256)
void gemm_bf16_wmma(const uint16_t* __restrict__ A, int lda,
                    const uint16_t* __restrict__ Bw, int ldb,
                    float* __restrict__ C, int ldc,
                    const float* __restrict__ bias,
                    int M, int N, int K, int accum) {
    static_assert(WGM * WGN == 8, "8 waves per block");
    static_assert(BM == WGM * WTM * 16 && BN == WGN * WTN * 16, "tile mismatch");
    (void)M; (void)N;

    __shared__ uint32_t shA[2][BM * 20];   // 20-dword (80B) pitch: conflict-free b128 gathers
    __shared__ uint32_t shB[2][BN * 20];

    const int tid  = threadIdx.x;
    const int lane = tid & 31;
    const int wave = tid >> 5;
    const int wm   = wave % WGM;
    const int wn   = wave / WGM;

    const long long blockRow = (long long)blockIdx.x * BM;
    const long long blockCol = (long long)blockIdx.y * BN;

    // per-lane fragment addressing (ISA 16-bit A layout, wave32)
    const int r      = lane & 15;
    const int dwBase = (lane >> 4) * 4;   // lane-half: K 0..7/16..23 vs 8..15/24..31

    v8f acc[WTM][WTN];
    for (int i = 0; i < WTM; ++i)
        for (int j = 0; j < WTN; ++j)
            for (int e = 0; e < 8; ++e) acc[i][j][e] = 0.f;

#if USE_ASYNC_LDS
    // -------- async DMA staging + double buffering (ASYNCcnt) --------
    auto stage_async = [&](int k0, int buf) {
        for (int c = tid; c < BM * 4; c += 256) {
            int row = c >> 2, col = c & 3;
            async_copy16(A + (blockRow + row) * (long long)lda + k0 + col * 8,
                         &shA[buf][row * 20 + col * 4]);
        }
        for (int c = tid; c < BN * 4; c += 256) {
            int row = c >> 2, col = c & 3;
            async_copy16(Bw + (blockCol + row) * (long long)ldb + k0 + col * 8,
                         &shB[buf][row * 20 + col * 4]);
        }
    };

    const int nIter = K / 32;
    stage_async(0, 0);
    for (int it = 0; it < nIter; ++it) {
        const int cur = it & 1;
        __builtin_amdgcn_s_wait_asynccnt(0);   // own copies into shX[cur] complete
        __syncthreads();                       // everyone's copies done; prev readers done
        if (it + 1 < nIter) stage_async((it + 1) * 32, cur ^ 1);

        v16bf aF[WTM], bF[WTN];
        for (int i = 0; i < WTM; ++i)
            aF[i] = make_frag(shA[cur], wm * WTM * 16 + i * 16 + r, dwBase);
        for (int j = 0; j < WTN; ++j)
            bF[j] = make_frag(shB[cur], wn * WTN * 16 + j * 16 + r, dwBase);
        for (int i = 0; i < WTM; ++i)
            for (int j = 0; j < WTN; ++j)
                acc[i][j] = __builtin_amdgcn_wmma_f32_16x16x32_bf16(
                    false, aF[i], false, bF[j], (short)0, acc[i][j], false, false);
        __syncthreads();   // protect shX[cur] before iteration it+2 overwrites it
    }
#else
    // -------- synchronous staging fallback --------
    for (int k0 = 0; k0 < K; k0 += 32) {
        __syncthreads();
        for (int c = tid; c < BM * 4; c += 256) {
            int row = c >> 2, col = c & 3;
            uint4 val = *reinterpret_cast<const uint4*>(A + (blockRow + row) * (long long)lda + k0 + col * 8);
            *reinterpret_cast<uint4*>(&shA[0][row * 20 + col * 4]) = val;
        }
        for (int c = tid; c < BN * 4; c += 256) {
            int row = c >> 2, col = c & 3;
            uint4 val = *reinterpret_cast<const uint4*>(Bw + (blockCol + row) * (long long)ldb + k0 + col * 8);
            *reinterpret_cast<uint4*>(&shB[0][row * 20 + col * 4]) = val;
        }
        if (k0 + 32 < K)  // hint next A tile into cache -> global_prefetch_b8
            __builtin_prefetch(A + (blockRow + (tid >> 2)) * (long long)lda + k0 + 32, 0, 1);
        __syncthreads();

        v16bf aF[WTM], bF[WTN];
        for (int i = 0; i < WTM; ++i)
            aF[i] = make_frag(shA[0], wm * WTM * 16 + i * 16 + r, dwBase);
        for (int j = 0; j < WTN; ++j)
            bF[j] = make_frag(shB[0], wn * WTN * 16 + j * 16 + r, dwBase);

        for (int i = 0; i < WTM; ++i)
            for (int j = 0; j < WTN; ++j) {
#if defined(__HIP_DEVICE_COMPILE__)
                acc[i][j] = __builtin_amdgcn_wmma_f32_16x16x32_bf16(
                    false, aF[i], false, bF[j], (short)0, acc[i][j], false, false);
#else
                (void)aF; (void)bF;
#endif
            }
    }
#endif

    // epilogue: C/D layout -> VGPR e, lanes 0-15: M=e,N=lane ; lanes 16-31: M=e+8,N=lane-16
    const int half  = lane >> 4;
    const int col16 = lane & 15;
    for (int i = 0; i < WTM; ++i)
        for (int j = 0; j < WTN; ++j) {
            long long gc = blockCol + wn * WTN * 16 + j * 16 + col16;
            float bv = bias ? bias[gc] : 0.f;
            for (int e = 0; e < 8; ++e) {
                long long gr = blockRow + wm * WTM * 16 + i * 16 + e + half * 8;
                long long o  = gr * (long long)ldc + gc;
                float val = acc[i][j][e] + bv;
                C[o] = accum ? (C[o] + val) : val;
            }
        }
}

// ---------------- fp32 -> bf16 convert ----------------
__global__ void cvt_bf16_kernel(const float* __restrict__ src, uint16_t* __restrict__ dst, long long n) {
    long long i = (long long)blockIdx.x * blockDim.x + threadIdx.x;
    long long stride = (long long)gridDim.x * blockDim.x;
    for (; i < n; i += stride) dst[i] = f2bf(src[i]);
}

// ---------------- pack [big;bfg] per layer into 16-wide bias ----------------
__global__ void pack_gbias_kernel(const float* __restrict__ big, const float* __restrict__ bfg,
                                  float* __restrict__ bg) {
    int i = threadIdx.x;               // 64 = kL*16
    int l = i >> 4, j = i & 15;
    bg[l * 16 + j] = (j < 8) ? big[l * 8 + j] : bfg[l * 8 + (j - 8)];
}

// ---------------- LayerNorm (D=512) -> bf16, wave per row ----------------
__global__ void ln_bf16_kernel(const float* __restrict__ h, const float* __restrict__ w,
                               uint16_t* __restrict__ out, long long rows) {
    long long row = (long long)blockIdx.x * (blockDim.x >> 5) + (threadIdx.x >> 5);
    if (row >= rows) return;
    int lane = threadIdx.x & 31;
    const float* r = h + row * kD;
    float vals[16], s = 0.f, ss = 0.f;
    for (int i = 0; i < 16; ++i) {
        float x = r[lane + i * 32];
        vals[i] = x; s += x; ss += x * x;
    }
    for (int m = 16; m; m >>= 1) { s += __shfl_xor(s, m, 32); ss += __shfl_xor(ss, m, 32); }
    float mu  = s * (1.f / kD);
    float var = ss * (1.f / kD) - mu * mu;
    float inv = rsqrtf(var + 1e-5f);
    uint16_t* o = out + row * kD;
    for (int i = 0; i < 16; ++i) {
        int c = lane + i * 32;
        o[c] = f2bf((vals[i] - mu) * inv * w[c]);
    }
}

// ---------------- depthwise causal conv (K=4) + swish ----------------
__global__ void conv_swish_kernel(const float* __restrict__ xm, int ldx,
                                  const float* __restrict__ w, const float* __restrict__ b,
                                  float* __restrict__ xa, long long n) {
    long long idx = (long long)blockIdx.x * blockDim.x + threadIdx.x;
    if (idx >= n) return;
    long long tok = idx >> 10;
    int c = (int)(idx & 1023);
    int t = (int)(tok & (kT - 1));
    float acc = b[c];
    for (int dt = 0; dt < kKc; ++dt) {
        int tt = t + dt - (kKc - 1);
        if (tt >= 0) acc += xm[(tok + dt - (kKc - 1)) * (long long)ldx + c] * w[c * kKc + dt];
    }
    xa[idx] = swishf(acc);
}

// ---------------- headwise 4x4 block-diagonal projection ----------------
__global__ void headwise_kernel(const float* __restrict__ x, int ldx,
                                const float* __restrict__ W,
                                float* __restrict__ out, long long n) {
    long long idx = (long long)blockIdx.x * blockDim.x + threadIdx.x;
    if (idx >= n) return;
    long long tok = idx >> 8;
    int nb = (int)(idx & 255);
    const float* xr = x + tok * (long long)ldx + nb * 4;
    const float* w  = W + nb * 16;
    float x0 = xr[0], x1 = xr[1], x2 = xr[2], x3 = xr[3];
    float* o = out + tok * (long long)kI + nb * 4;
    for (int oo = 0; oo < 4; ++oo)
        o[oo] = x0 * w[oo * 4 + 0] + x1 * w[oo * 4 + 1] + x2 * w[oo * 4 + 2] + x3 * w[oo * 4 + 3];
}

// ---------------- pack gin = concat(q,k,v) -> bf16 ----------------
__global__ void gin_pack_kernel(const float* __restrict__ q, const float* __restrict__ k,
                                const float* __restrict__ v, uint16_t* __restrict__ gin,
                                long long n) {
    long long idx = (long long)blockIdx.x * blockDim.x + threadIdx.x;
    if (idx >= n) return;
    long long tok = idx / (3 * kI);
    int c = (int)(idx - tok * (3 * kI));
    float x;
    if (c < kI)          x = q[tok * kI + c];
    else if (c < 2 * kI) x = k[tok * kI + (c - kI)];
    else                 x = v[tok * kI + (c - 2 * kI)];
    gin[idx] = f2bf(x);
}

// ---------------- mLSTM cell + fused per-head layernorm ----------------
// one workgroup (256 threads) per (batch-sample, head)
__global__ __launch_bounds__(256)
void mlstm_cell_kernel(const float* __restrict__ q, const float* __restrict__ k,
                       const float* __restrict__ v, const float* __restrict__ gates,
                       const float* __restrict__ mhw, float* __restrict__ hn) {
    __shared__ float QS[kT][kDH + 1];        // later reused for hc
    __shared__ float KS[kT][kDH + 1];
    __shared__ float VS[kT][kDH + 1];
    __shared__ float CM[kT][kT + 1];
    __shared__ float DL[kT][kT + 1];
    __shared__ float cum[kT + 1], mx[kT], scl[kT], mean[kT], inv[kT], igS[kT];

    const int b = blockIdx.x >> 3;
    const int h = blockIdx.x & 7;
    const int tid = threadIdx.x;
    const long long base = ((long long)b * kT) * kI + h * kDH;

    // load q/k/v head slices (float4)
    for (int i = tid; i < kT * kDH / 4; i += 256) {
        int t = i >> 5, c4 = i & 31;
        long long o = base + (long long)t * kI + c4 * 4;
        float4 qv = *reinterpret_cast<const float4*>(q + o);
        float4 kv = *reinterpret_cast<const float4*>(k + o);
        float4 vv = *reinterpret_cast<const float4*>(v + o);
        QS[t][c4 * 4 + 0] = qv.x; QS[t][c4 * 4 + 1] = qv.y; QS[t][c4 * 4 + 2] = qv.z; QS[t][c4 * 4 + 3] = qv.w;
        KS[t][c4 * 4 + 0] = kv.x; KS[t][c4 * 4 + 1] = kv.y; KS[t][c4 * 4 + 2] = kv.z; KS[t][c4 * 4 + 3] = kv.w;
        VS[t][c4 * 4 + 0] = vv.x; VS[t][c4 * 4 + 1] = vv.y; VS[t][c4 * 4 + 2] = vv.z; VS[t][c4 * 4 + 3] = vv.w;
    }
    if (tid < kT) igS[tid] = gates[((long long)b * kT + tid) * 16 + h];
    if (tid == 0) {
        cum[0] = 0.f;
        for (int s = 0; s < kT; ++s) {
            float x  = gates[((long long)b * kT + s) * 16 + 8 + h];
            float lf = fminf(x, 0.f) - log1pf(expf(-fabsf(x)));   // log_sigmoid
            cum[s + 1] = cum[s] + lf;
        }
    }
    __syncthreads();

    // scores + log-gate matrix
    const float invs = 0.08838834764831845f;   // 1/sqrt(128)
    for (int i = tid; i < kT * kT; i += 256) {
        int t = i >> 5, s = i & 31;
        float acc = 0.f;
        for (int d = 0; d < kDH; ++d) acc += QS[t][d] * KS[s][d];
        CM[t][s] = acc * invs;
        DL[t][s] = (s <= t) ? (cum[t + 1] - cum[s + 1] + igS[s]) : -__builtin_inff();
    }
    __syncthreads();

    // per-row max, gate, normalize factor
    if (tid < kT) {
        int t = tid;
        float m = -__builtin_inff();
        for (int s = 0; s <= t; ++s) m = fmaxf(m, DL[t][s]);
        mx[t] = m;
        float sum = 0.f;
        for (int s = 0; s < kT; ++s) {
            float c = (s <= t) ? CM[t][s] * expf(DL[t][s] - m) : 0.f;
            CM[t][s] = c;
            sum += c;
        }
        float nrm = fmaxf(fabsf(sum), expf(-m));
        scl[t] = 1.f / (nrm + 1e-6f);
    }
    __syncthreads();

    // hc = (CM * scl) @ V  (write into QS, q no longer needed)
    for (int i = tid; i < kT * kDH; i += 256) {
        int t = i >> 7, d = i & 127;
        float acc = 0.f;
        for (int s = 0; s <= t; ++s) acc += CM[t][s] * VS[s][d];
        QS[t][d] = acc * scl[t];
    }
    __syncthreads();

    // per-head layernorm stats
    if (tid < kT) {
        float s = 0.f, ss = 0.f;
        for (int d = 0; d < kDH; ++d) { float x = QS[tid][d]; s += x; ss += x * x; }
        float mu  = s * (1.f / kDH);
        float var = ss * (1.f / kDH) - mu * mu;
        mean[tid] = mu;
        inv[tid]  = rsqrtf(var + 1e-5f);
    }
    __syncthreads();

    // write hn (b, t, h*DH + d) with mh weight
    for (int i = tid; i < kT * kDH; i += 256) {
        int t = i >> 7, d = i & 127;
        hn[((long long)b * kT + t) * kI + h * kDH + d] =
            (QS[t][d] - mean[t]) * inv[t] * mhw[h * kDH + d];
    }
}

// ---------------- combine: out = (hn + skip*xa) * swish(z) -> bf16 ----------------
__global__ void combine_kernel(const float* __restrict__ hn, const float* __restrict__ xa,
                               const float* __restrict__ up, const float* __restrict__ skip,
                               uint16_t* __restrict__ outbf, long long n) {
    long long idx = (long long)blockIdx.x * blockDim.x + threadIdx.x;
    if (idx >= n) return;
    long long tok = idx >> 10;
    int c = (int)(idx & 1023);
    float z = up[tok * (2LL * kI) + kI + c];
    float val = (hn[idx] + skip[c] * xa[idx]) * swishf(z);
    outbf[idx] = f2bf(val);
}

// ---------------- post-LN + 3-layer GELU MLP on last timestep ----------------
__device__ __forceinline__ float geluf(float x) {
    return 0.5f * x * (1.f + erff(x * 0.70710678118654752f));
}

__global__ __launch_bounds__(64)
void final_mlp_kernel(const float* __restrict__ h, const float* __restrict__ post_w,
                      const float* __restrict__ W1, const float* __restrict__ b1,
                      const float* __restrict__ W2, const float* __restrict__ b2,
                      const float* __restrict__ W3, const float* __restrict__ b3,
                      float* __restrict__ out) {
    __shared__ float row[kD], y1[64], y2[32], sr[64], ssr[64], mu_s, inv_s;
    const int r   = blockIdx.x;                       // 0..B-1
    const int tid = threadIdx.x;
    const float* hr = h + ((long long)r * kT + (kT - 1)) * kD;

    float s = 0.f, ss = 0.f;
    for (int i = tid; i < kD; i += 64) { float x = hr[i]; row[i] = x; s += x; ss += x * x; }
    sr[tid] = s; ssr[tid] = ss;
    __syncthreads();
    if (tid == 0) {
        float a = 0.f, bb = 0.f;
        for (int i = 0; i < 64; ++i) { a += sr[i]; bb += ssr[i]; }
        float mu  = a * (1.f / kD);
        float var = bb * (1.f / kD) - mu * mu;
        mu_s = mu; inv_s = rsqrtf(var + 1e-5f);
    }
    __syncthreads();
    for (int i = tid; i < kD; i += 64) row[i] = (row[i] - mu_s) * inv_s * post_w[i];
    __syncthreads();
    {
        float acc = b1[tid];
        const float* w = W1 + tid * kD;
        for (int d = 0; d < kD; ++d) acc += row[d] * w[d];
        y1[tid] = geluf(acc);
    }
    __syncthreads();
    if (tid < 32) {
        float acc = b2[tid];
        const float* w = W2 + tid * 64;
        for (int d = 0; d < 64; ++d) acc += y1[d] * w[d];
        y2[tid] = geluf(acc);
    }
    __syncthreads();
    if (tid < 16) {
        float acc = b3[tid];
        const float* w = W3 + tid * 32;
        for (int d = 0; d < 32; ++d) acc += y2[d] * w[d];
        out[(long long)r * 16 + tid] = acc;
    }
}

// ---------------- host orchestration ----------------
static inline int eg(long long n) { return (int)((n + 255) / 256); }

extern "C" void kernel_launch(void* const* d_in, const int* in_sizes, int n_in,
                              void* d_out, int out_size, void* d_ws, size_t ws_size,
                              hipStream_t stream) {
    (void)in_sizes; (void)n_in; (void)out_size; (void)ws_size;

    const float* x      = (const float*)d_in[0];
    const float* Wi     = (const float*)d_in[1];
    const float* bi     = (const float*)d_in[2];
    const float* ln_w   = (const float*)d_in[3];
    const float* Wup    = (const float*)d_in[4];
    const float* conv_w = (const float*)d_in[5];
    const float* conv_b = (const float*)d_in[6];
    const float* Wq     = (const float*)d_in[7];
    const float* Wk     = (const float*)d_in[8];
    const float* Wv     = (const float*)d_in[9];
    const float* Wig    = (const float*)d_in[10];
    const float* big    = (const float*)d_in[11];
    const float* Wfg    = (const float*)d_in[12];
    const float* bfg    = (const float*)d_in[13];
    const float* skip   = (const float*)d_in[14];
    const float* mhln_w = (const float*)d_in[15];
    const float* Wdown  = (const float*)d_in[16];
    const float* post_w = (const float*)d_in[17];
    const float* W1     = (const float*)d_in[18];
    const float* b1     = (const float*)d_in[19];
    const float* W2     = (const float*)d_in[20];
    const float* b2     = (const float*)d_in[21];
    const float* W3     = (const float*)d_in[22];
    const float* b3     = (const float*)d_in[23];
    float* out = (float*)d_out;

    char* ws = (char*)d_ws;
    size_t off = 0;
    auto take = [&](size_t bytes) { size_t o = off; off += (bytes + 255) & ~(size_t)255; return o; };

    uint16_t* xs_bf   = (uint16_t*)(ws + take(kNT * kFPS * 2));
    float*    h       = (float*)   (ws + take(kNT * kD * 4));
    uint16_t* hln_bf  = (uint16_t*)(ws + take(kNT * kD * 2));
    float*    up      = (float*)   (ws + take(kNT * 2 * kI * 4));
    float*    xa      = (float*)   (ws + take(kNT * kI * 4));
    float*    qb      = (float*)   (ws + take(kNT * kI * 4));
    float*    kb      = (float*)   (ws + take(kNT * kI * 4));
    float*    vb      = (float*)   (ws + take(kNT * kI * 4));
    uint16_t* gin_bf  = (uint16_t*)(ws + take(kNT * 3 * kI * 2));
    float*    gates   = (float*)   (ws + take(kNT * 16 * 4));
    float*    hn      = (float*)   (ws + take(kNT * kI * 4));
    uint16_t* out_bf  = (uint16_t*)(ws + take(kNT * kI * 2));
    uint16_t* Wi_bf   = (uint16_t*)(ws + take((size_t)kD * kFPS * 2));
    uint16_t* Wup_bf  = (uint16_t*)(ws + take((size_t)kL * 2 * kI * kD * 2));
    uint16_t* Wdn_bf  = (uint16_t*)(ws + take((size_t)kL * kD * kI * 2));
    uint16_t* Wg_bf   = (uint16_t*)(ws + take((size_t)kL * 16 * 3 * kI * 2));
    float*    bg      = (float*)   (ws + take((size_t)kL * 16 * 4));

    // --- weight / input conversion (deterministic, every call) ---
    cvt_bf16_kernel<<<eg(kNT * kFPS), 256, 0, stream>>>(x, xs_bf, kNT * kFPS);   // T*FPS == INPUT_DIM
    cvt_bf16_kernel<<<eg((long long)kD * kFPS), 256, 0, stream>>>(Wi, Wi_bf, (long long)kD * kFPS);
    cvt_bf16_kernel<<<eg((long long)kL * 2 * kI * kD), 256, 0, stream>>>(Wup, Wup_bf, (long long)kL * 2 * kI * kD);
    cvt_bf16_kernel<<<eg((long long)kL * kD * kI), 256, 0, stream>>>(Wdown, Wdn_bf, (long long)kL * kD * kI);
    for (int l = 0; l < kL; ++l) {   // pack [Wig;Wfg] -> (16, 3I) per layer
        cvt_bf16_kernel<<<eg((long long)kNH * 3 * kI), 256, 0, stream>>>(
            Wig + (long long)l * kNH * 3 * kI, Wg_bf + (long long)l * 16 * 3 * kI, (long long)kNH * 3 * kI);
        cvt_bf16_kernel<<<eg((long long)kNH * 3 * kI), 256, 0, stream>>>(
            Wfg + (long long)l * kNH * 3 * kI, Wg_bf + (long long)l * 16 * 3 * kI + (long long)kNH * 3 * kI,
            (long long)kNH * 3 * kI);
    }
    pack_gbias_kernel<<<1, 64, 0, stream>>>(big, bfg, bg);

    // --- input projection: h = xs @ Wi^T + bi   (M=16384, N=512, K=64) ---
    {
        dim3 grid((unsigned)(kNT / 128), kD / 128);
        gemm_bf16_wmma<128, 128, 2, 4, 4, 2><<<grid, 256, 0, stream>>>(
            xs_bf, kFPS, Wi_bf, kFPS, h, kD, bi, (int)kNT, kD, kFPS, 0);
    }

    // --- L blocks ---
    for (int l = 0; l < kL; ++l) {
        ln_bf16_kernel<<<(unsigned)(kNT / 8), 256, 0, stream>>>(h, ln_w + (long long)l * kD, hln_bf, kNT);

        { // up = ln(h) @ Wup^T   (N=2048, K=512)
            dim3 grid((unsigned)(kNT / 128), 2 * kI / 128);
            gemm_bf16_wmma<128, 128, 2, 4, 4, 2><<<grid, 256, 0, stream>>>(
                hln_bf, kD, Wup_bf + (long long)l * 2 * kI * kD, kD, up, 2 * kI, nullptr,
                (int)kNT, 2 * kI, kD, 0);
        }

        conv_swish_kernel<<<eg(kNT * kI), 256, 0, stream>>>(
            up, 2 * kI, conv_w + (long long)l * kI * kKc, conv_b + (long long)l * kI, xa, kNT * kI);

        headwise_kernel<<<eg(kNT * kNB), 256, 0, stream>>>(xa, kI, Wq + (long long)l * kNB * 16, qb, kNT * kNB);
        headwise_kernel<<<eg(kNT * kNB), 256, 0, stream>>>(xa, kI, Wk + (long long)l * kNB * 16, kb, kNT * kNB);
        headwise_kernel<<<eg(kNT * kNB), 256, 0, stream>>>(up, 2 * kI, Wv + (long long)l * kNB * 16, vb, kNT * kNB);

        gin_pack_kernel<<<eg(kNT * 3 * kI), 256, 0, stream>>>(qb, kb, vb, gin_bf, kNT * 3 * kI);

        { // gates = gin @ [Wig;Wfg]^T + [big;bfg]   (N=16, K=3072)
            dim3 grid((unsigned)(kNT / 256), 1);
            gemm_bf16_wmma<256, 16, 8, 1, 2, 1><<<grid, 256, 0, stream>>>(
                gin_bf, 3 * kI, Wg_bf + (long long)l * 16 * 3 * kI, 3 * kI, gates, 16,
                bg + (long long)l * 16, (int)kNT, 16, 3 * kI, 0);
        }

        mlstm_cell_kernel<<<(unsigned)(kB * kNH), 256, 0, stream>>>(
            qb, kb, vb, gates, mhln_w + (long long)l * kI, hn);

        combine_kernel<<<eg(kNT * kI), 256, 0, stream>>>(hn, xa, up, skip + (long long)l * kI, out_bf, kNT * kI);

        { // h += out @ Wdown^T   (N=512, K=1024)
            dim3 grid((unsigned)(kNT / 128), kD / 128);
            gemm_bf16_wmma<128, 128, 2, 4, 4, 2><<<grid, 256, 0, stream>>>(
                out_bf, kI, Wdn_bf + (long long)l * kD * kI, kI, h, kD, nullptr,
                (int)kNT, kD, kI, 1);
        }
    }

    // --- head: post-LN on last timestep + MLP ---
    final_mlp_kernel<<<kB, 64, 0, stream>>>(h, post_w, W1, b1, W2, b2, W3, b3, out);
}
